// GPTLM_15247133900804
// MI455X (gfx1250) — matrix operand order repaired and checked by reading, not compiled, not run on hardware
//
#include <hip/hip_runtime.h>

// ---------------- problem constants ----------------
#define Bv   64
#define Tv   256
#define Cv   384
#define Hv   6
#define HSv  64
#define Lv   6
#define Vv   65
#define DFFv 1536
#define BTv  (Bv * Tv)   // 16384

typedef __attribute__((ext_vector_type(16))) __bf16 v16bf;
typedef __attribute__((ext_vector_type(8)))  float  v8f;
typedef __attribute__((ext_vector_type(4)))  unsigned int v4u;

// ---------------- bf16 helpers (storage = unsigned short) ----------------
__device__ __forceinline__ unsigned short f2b(float f) {
    unsigned int u = __float_as_uint(f);
    u += 0x7fffu + ((u >> 16) & 1u);          // round-to-nearest-even
    return (unsigned short)(u >> 16);
}
__device__ __forceinline__ float b2f(unsigned short h) {
    return __uint_as_float(((unsigned int)h) << 16);
}

union FragU { v16bf v; v4u q[2]; unsigned int u[8]; };

// ---------------- epilogue helper (inlined; stores one 16x16 tile) ----------------
template <bool BIAS, bool RELU, bool OUTBF, bool TRANSC>
__device__ __forceinline__ void store_tile(const v8f& acc, int mt, int hi, int col,
                                           int N, int ldc, long long cOff,
                                           const float* __restrict__ bias,
                                           void* __restrict__ Cout, float scale)
{
    if (col < N) {
        const float bv = BIAS ? bias[col] : 0.f;
#pragma unroll
        for (int r = 0; r < 8; ++r) {
            const int row = mt * 16 + r + hi * 8;
            float val = acc[r] * scale + bv;
            if (RELU) val = fmaxf(val, 0.f);
            if (TRANSC) {
                // V projection: out[b*C*T + col*T + t], row = b*T + t (T=256)
                const long long o = (long long)(row >> 8) * (Cv * Tv)
                                  + (long long)col * Tv + (row & (Tv - 1));
                ((unsigned short*)Cout)[o] = f2b(val);
            } else if (OUTBF) {
                ((unsigned short*)Cout)[cOff + (long long)row * ldc + col] = f2b(val);
            } else {
                ((float*)Cout)[cOff + (long long)row * ldc + col] = val;
            }
        }
    }
}

// ---------------- generic bf16 WMMA GEMM ----------------
// C[M,N] = act( scale * A[M,K] x B[K,N] + bias )
//   A: bf16 row-major [M,K], lda (16B-aligned rows)
//   B: bf16 K-CONTIGUOUS: element (k,n) at Bt[n*ldb + k]
// Each wave computes a 16x64 strip (4 n-tiles, 4 independent accumulators).
// Inner loop is branch-free: out-of-range columns load from a clamped
// address and are simply not stored. CAUSAL skips whole groups above the
// diagonal (wave-uniform, before the loop); in-group garbage above the
// diagonal is masked later by the causal softmax.
// batch z: operand offset = (z/hd)*s?1 + (z%hd)*s?2
template <bool BIAS, bool RELU, bool OUTBF, bool CAUSAL, bool TRANSC>
__global__ void k_gemm(const unsigned short* __restrict__ A,
                       const unsigned short* __restrict__ Bt,
                       const float* __restrict__ bias,
                       void* __restrict__ Cout,
                       int M, int N, int K, int lda, int ldb, int ldc,
                       int hd,
                       long long sA1, long long sA2,
                       long long sB1, long long sB2,
                       long long sC1, long long sC2,
                       float scale)
{
    const int wave = threadIdx.x >> 5;     // blockDim.x == 128 -> 4 waves
    const int lane = threadIdx.x & 31;
    const int tN  = (N + 15) >> 4;
    const int tM  = M >> 4;                // M always a multiple of 16 here
    const int tNG = (tN + 3) >> 2;         // groups of 4 n-tiles
    const int tile = blockIdx.x * 4 + wave;
    if (tile >= tM * tNG) return;          // wave-uniform
    const int mt  = tile / tNG;
    const int ntg = tile % tNG;
    if (CAUSAL && ntg * 4 > mt) return;    // wave-uniform

    const int z = blockIdx.z;
    const unsigned short* Ab = A  + (long long)(z / hd) * sA1 + (long long)(z % hd) * sA2;
    const unsigned short* Bb = Bt + (long long)(z / hd) * sB1 + (long long)(z % hd) * sB2;
    const long long cOff = (long long)(z / hd) * sC1 + (long long)(z % hd) * sC2;

    const int hi = lane >> 4;              // 0 or 1
    const int lr = lane & 15;
    const int arow = mt * 16 + lr;
    const int col0 = ntg * 64 + lr;        // columns of the 4 tiles: col0 + {0,16,32,48}

    // clamped columns for load addressing (loads always in-bounds)
    const int Nm1 = N - 1;
    const int c0 = (col0      < N) ? col0      : Nm1;
    const int c1 = (col0 + 16 < N) ? col0 + 16 : Nm1;
    const int c2 = (col0 + 32 < N) ? col0 + 32 : Nm1;
    const int c3 = (col0 + 48 < N) ? col0 + 48 : Nm1;

    const unsigned short* arp  = Ab + (long long)arow * lda + hi * 16;
    const unsigned short* brp0 = Bb + (long long)c0 * ldb + hi * 16;
    const unsigned short* brp1 = Bb + (long long)c1 * ldb + hi * 16;
    const unsigned short* brp2 = Bb + (long long)c2 * ldb + hi * 16;
    const unsigned short* brp3 = Bb + (long long)c3 * ldb + hi * 16;

    v8f acc0 = {0.f,0.f,0.f,0.f,0.f,0.f,0.f,0.f};
    v8f acc1 = acc0, acc2 = acc0, acc3 = acc0;

    for (int k0 = 0; k0 < K; k0 += 32) {
        FragU a, b0, b1, b2, b3;
        // A fragment: 8 contiguous dwords (32 bytes)
        a.q[0]  = *(const v4u*)(arp);
        a.q[1]  = *(const v4u*)(arp + 8);
        // B fragments: 8 contiguous dwords (32 bytes)
        b0.q[0] = *(const v4u*)(brp0);
        b0.q[1] = *(const v4u*)(brp0 + 8);
        b1.q[0] = *(const v4u*)(brp1);
        b1.q[1] = *(const v4u*)(brp1 + 8);
        b2.q[0] = *(const v4u*)(brp2);
        b2.q[1] = *(const v4u*)(brp2 + 8);
        b3.q[0] = *(const v4u*)(brp3);
        b3.q[1] = *(const v4u*)(brp3 + 8);

        acc0 = __builtin_amdgcn_wmma_f32_16x16x32_bf16(false, a.v, false, b0.v, (short)0, acc0, false, false);
        acc1 = __builtin_amdgcn_wmma_f32_16x16x32_bf16(false, a.v, false, b1.v, (short)0, acc1, false, false);
        acc2 = __builtin_amdgcn_wmma_f32_16x16x32_bf16(false, a.v, false, b2.v, (short)0, acc2, false, false);
        acc3 = __builtin_amdgcn_wmma_f32_16x16x32_bf16(false, a.v, false, b3.v, (short)0, acc3, false, false);

        arp += 32; brp0 += 32; brp1 += 32; brp2 += 32; brp3 += 32;
    }

    store_tile<BIAS,RELU,OUTBF,TRANSC>(acc0, mt, hi, col0,      N, ldc, cOff, bias, Cout, scale);
    store_tile<BIAS,RELU,OUTBF,TRANSC>(acc1, mt, hi, col0 + 16, N, ldc, cOff, bias, Cout, scale);
    store_tile<BIAS,RELU,OUTBF,TRANSC>(acc2, mt, hi, col0 + 32, N, ldc, cOff, bias, Cout, scale);
    store_tile<BIAS,RELU,OUTBF,TRANSC>(acc3, mt, hi, col0 + 48, N, ldc, cOff, bias, Cout, scale);
}

// ---------------- embedding: x = tok_emb[idx] + pos_emb ----------------
__global__ void k_embed(const int* __restrict__ idx,
                        const float* __restrict__ tok,
                        const float* __restrict__ pos,
                        float* __restrict__ xf,
                        unsigned short* __restrict__ xb)
{
    int i = blockIdx.x * 256 + threadIdx.x;
    if (i >= BTv * Cv) return;
    int c  = i % Cv;
    int bt = i / Cv;
    int t  = bt % Tv;
    float v = tok[(long long)idx[bt] * Cv + c] + pos[(long long)t * Cv + c];
    xf[i] = v;
    xb[i] = f2b(v);
}

// ---------------- weight conversions (f32 -> bf16, K-contiguous) ----------------
// (L,H,C,HS) -> out[l][h*HS+d][c]
__global__ void k_cvt_qkv_t(const float* __restrict__ in, unsigned short* __restrict__ out)
{
    int i = blockIdx.x * 256 + threadIdx.x;
    if (i >= Lv * Cv * Cv) return;
    int l = i / (Cv * Cv);
    int r = i % (Cv * Cv);
    int j = r / Cv;           // output row  = h*HS + d
    int c = r % Cv;           // output col  = k index
    int h = j / HSv;
    int d = j % HSv;
    out[i] = f2b(in[(((long long)l * Hv + h) * Cv + c) * HSv + d]);
}
// batched transpose: in [batch][R][Cc] -> out [batch][Cc][R]
__global__ void k_cvt_t(const float* __restrict__ in, unsigned short* __restrict__ out,
                        int batch, int R, int Cc)
{
    int i = blockIdx.x * 256 + threadIdx.x;
    if (i >= batch * R * Cc) return;
    int b  = i / (R * Cc);
    int r2 = i % (R * Cc);
    int n  = r2 / R;
    int k  = r2 % R;
    out[i] = f2b(in[(long long)b * R * Cc + (long long)k * Cc + n]);
}

// ---------------- causal softmax over rows of S (bf16 in-place) ----------------
__global__ void k_softmax(unsigned short* __restrict__ S)
{
    const int row = blockIdx.x;            // B*H*T rows
    const int t   = row % Tv;
    const int s   = threadIdx.x;           // 256 threads = row length
    const long long base = (long long)row * Tv;
    __shared__ float red[256];

    const bool ok = (s <= t);
    float v = ok ? b2f(S[base + s]) : -1e30f;
    red[s] = v; __syncthreads();
    for (int k = 128; k > 0; k >>= 1) { if (s < k) red[s] = fmaxf(red[s], red[s + k]); __syncthreads(); }
    const float mx = red[0]; __syncthreads();
    float e = ok ? __expf(v - mx) : 0.f;
    red[s] = e; __syncthreads();
    for (int k = 128; k > 0; k >>= 1) { if (s < k) red[s] += red[s + k]; __syncthreads(); }
    const float inv = 1.f / red[0];
    S[base + s] = f2b(e * inv);
}

// ---------------- layernorm (+ optional residual add) ----------------
template <bool ADD>
__global__ void k_ln(const float* __restrict__ in,
                     const float* __restrict__ g,
                     const float* __restrict__ be,
                     float* __restrict__ resid,
                     unsigned short* __restrict__ outb)
{
    const int row = blockIdx.x;            // BT rows
    const int tid = threadIdx.x;           // 256 threads, C=384 -> <=2 elems/thread
    const float* xr = in + (long long)row * Cv;
    __shared__ float red[256];

    float v0 = xr[tid];
    float v1 = (tid < Cv - 256) ? xr[tid + 256] : 0.f;
    red[tid] = v0 + v1; __syncthreads();
    for (int k = 128; k > 0; k >>= 1) { if (tid < k) red[tid] += red[tid + k]; __syncthreads(); }
    const float mu = red[0] / (float)Cv; __syncthreads();

    float d0 = v0 - mu;
    float d1 = (tid < Cv - 256) ? (v1 - mu) : 0.f;
    red[tid] = d0 * d0 + d1 * d1; __syncthreads();
    for (int k = 128; k > 0; k >>= 1) { if (tid < k) red[tid] += red[tid + k]; __syncthreads(); }
    const float inv = rsqrtf(red[0] / (float)Cv + 1e-5f);

    {
        const int c = tid;
        float y = d0 * inv * g[c] + be[c];
        const long long o = (long long)row * Cv + c;
        if (ADD) { float nx = resid[o] + y; resid[o] = nx; outb[o] = f2b(nx); }
        else     { outb[o] = f2b(y); }
    }
    if (tid < Cv - 256) {
        const int c = tid + 256;
        float y = d1 * inv * g[c] + be[c];
        const long long o = (long long)row * Cv + c;
        if (ADD) { float nx = resid[o] + y; resid[o] = nx; outb[o] = f2b(nx); }
        else     { outb[o] = f2b(y); }
    }
    (void)resid;
}

// ---------------- cross-entropy per row ----------------
__global__ void k_rowloss(const float* __restrict__ logits,
                          const int* __restrict__ target,
                          float* __restrict__ rowloss)
{
    const int row = blockIdx.x;            // BT rows
    const int tid = threadIdx.x;           // 128 threads, V=65
    __shared__ float red[128];

    float v = (tid < Vv) ? logits[(long long)row * Vv + tid] : -1e30f;
    red[tid] = v; __syncthreads();
    for (int k = 64; k > 0; k >>= 1) { if (tid < k) red[tid] = fmaxf(red[tid], red[tid + k]); __syncthreads(); }
    const float mx = red[0]; __syncthreads();
    float e = (tid < Vv) ? __expf(v - mx) : 0.f;
    red[tid] = e; __syncthreads();
    for (int k = 64; k > 0; k >>= 1) { if (tid < k) red[tid] += red[tid + k]; __syncthreads(); }
    if (tid == 0) {
        const int tg = target[row];
        rowloss[row] = -(logits[(long long)row * Vv + tg] - mx - __logf(red[0]));
    }
}

// deterministic final reduction (fixed order, no atomics)
__global__ void k_loss_reduce(const float* __restrict__ rowloss, float* __restrict__ out)
{
    const int tid = threadIdx.x;           // 256
    __shared__ float red[256];
    float s = 0.f;
    for (int i = tid; i < BTv; i += 256) s += rowloss[i];
    red[tid] = s; __syncthreads();
    for (int k = 128; k > 0; k >>= 1) { if (tid < k) red[tid] += red[tid + k]; __syncthreads(); }
    if (tid == 0) out[0] = red[0] / (float)BTv;
}

// ---------------- host-side orchestration ----------------
extern "C" void kernel_launch(void* const* d_in, const int* in_sizes, int n_in,
                              void* d_out, int out_size, void* d_ws, size_t ws_size,
                              hipStream_t stream)
{
    (void)in_sizes; (void)n_in; (void)out_size; (void)ws_size;

    const int*   idx    = (const int*)  d_in[0];
    const int*   target = (const int*)  d_in[1];
    const float* tok    = (const float*)d_in[2];
    const float* pos    = (const float*)d_in[3];
    const float* wq     = (const float*)d_in[4];
    const float* wk     = (const float*)d_in[5];
    const float* wv     = (const float*)d_in[6];
    const float* wproj  = (const float*)d_in[7];
    const float* bproj  = (const float*)d_in[8];
    const float* ln1g   = (const float*)d_in[9];
    const float* ln1b   = (const float*)d_in[10];
    const float* w1     = (const float*)d_in[11];
    const float* b1     = (const float*)d_in[12];
    const float* w2     = (const float*)d_in[13];
    const float* b2     = (const float*)d_in[14];
    const float* ln2g   = (const float*)d_in[15];
    const float* ln2b   = (const float*)d_in[16];
    const float* lnfg   = (const float*)d_in[17];
    const float* lnfb   = (const float*)d_in[18];
    const float* wlm    = (const float*)d_in[19];
    const float* blm    = (const float*)d_in[20];

    float* out = (float*)d_out;            // [BT*V] logits + [1] loss
    char*  ws  = (char*)d_ws;

    // ---- workspace layout (bytes) ----
    const size_t oXF  = 0;                                 // f32  BT*C
    const size_t oXB  = oXF  + (size_t)BTv * Cv * 4;       // bf16 BT*C
    const size_t oQB  = oXB  + (size_t)BTv * Cv * 2;
    const size_t oKB  = oQB  + (size_t)BTv * Cv * 2;
    const size_t oVT  = oKB  + (size_t)BTv * Cv * 2;       // bf16 V transposed [B][C][T]
    const size_t oAT  = oVT  + (size_t)BTv * Cv * 2;       // attn-concat / lnf-out bf16
    const size_t oSA  = oAT  + (size_t)BTv * Cv * 2;       // f32 sublayer out
    const size_t oSC  = oSA  + (size_t)BTv * Cv * 4;       // bf16 scores == bf16 ffn hidden
    const size_t oWQ  = oSC  + (size_t)BTv * DFFv * 2;
    const size_t oWK  = oWQ  + (size_t)Lv * Cv * Cv * 2;
    const size_t oWV  = oWK  + (size_t)Lv * Cv * Cv * 2;
    const size_t oWP  = oWV  + (size_t)Lv * Cv * Cv * 2;
    const size_t oW1  = oWP  + (size_t)Lv * Cv * Cv * 2;
    const size_t oW2  = oW1  + (size_t)Lv * Cv * DFFv * 2;
    const size_t oWL  = oW2  + (size_t)Lv * DFFv * Cv * 2;
    const size_t oRL  = oWL  + (size_t)Cv * Vv * 2;        // f32 row losses

    float*          XF  = (float*)(ws + oXF);
    unsigned short* XB  = (unsigned short*)(ws + oXB);
    unsigned short* QB  = (unsigned short*)(ws + oQB);
    unsigned short* KB  = (unsigned short*)(ws + oKB);
    unsigned short* VT  = (unsigned short*)(ws + oVT);
    unsigned short* AT  = (unsigned short*)(ws + oAT);
    float*          SA  = (float*)(ws + oSA);
    unsigned short* SC  = (unsigned short*)(ws + oSC);
    unsigned short* WQt = (unsigned short*)(ws + oWQ);
    unsigned short* WKt = (unsigned short*)(ws + oWK);
    unsigned short* WVt = (unsigned short*)(ws + oWV);
    unsigned short* WPt = (unsigned short*)(ws + oWP);
    unsigned short* W1t = (unsigned short*)(ws + oW1);
    unsigned short* W2t = (unsigned short*)(ws + oW2);
    unsigned short* WLt = (unsigned short*)(ws + oWL);
    float*          RL  = (float*)(ws + oRL);

    // ---- weight conversion to bf16, K-contiguous (transposed) ----
    {
        const int nqkv = Lv * Cv * Cv;
        dim3 g((nqkv + 255) / 256);
        k_cvt_qkv_t<<<g, 256, 0, stream>>>(wq, WQt);
        k_cvt_qkv_t<<<g, 256, 0, stream>>>(wk, WKt);
        k_cvt_qkv_t<<<g, 256, 0, stream>>>(wv, WVt);
        k_cvt_t<<<g, 256, 0, stream>>>(wproj, WPt, Lv, Cv, Cv);
        const int nff = Lv * Cv * DFFv;
        k_cvt_t<<<(nff + 255) / 256, 256, 0, stream>>>(w1, W1t, Lv, Cv, DFFv);
        k_cvt_t<<<(nff + 255) / 256, 256, 0, stream>>>(w2, W2t, Lv, DFFv, Cv);
        const int nlm = Cv * Vv;
        k_cvt_t<<<(nlm + 255) / 256, 256, 0, stream>>>(wlm, WLt, 1, Cv, Vv);
    }

    // ---- embedding ----
    k_embed<<<(BTv * Cv + 255) / 256, 256, 0, stream>>>(idx, tok, pos, XF, XB);

    auto blocks_for = [](int M, int N) {
        int tN  = (N + 15) / 16;
        int tiles = (M / 16) * ((tN + 3) / 4);
        return (unsigned)((tiles + 3) / 4);
    };

    for (int l = 0; l < Lv; ++l) {
        const unsigned short* wql = WQt + (size_t)l * Cv * Cv;
        const unsigned short* wkl = WKt + (size_t)l * Cv * Cv;
        const unsigned short* wvl = WVt + (size_t)l * Cv * Cv;
        const unsigned short* wpl = WPt + (size_t)l * Cv * Cv;
        const unsigned short* w1l = W1t + (size_t)l * Cv * DFFv;
        const unsigned short* w2l = W2t + (size_t)l * DFFv * Cv;

        // Q/K = XB x W -> bf16 [BT, C] ; V -> transposed bf16 [B][C][T]
        {
            dim3 g(blocks_for(BTv, Cv), 1, 1);
            k_gemm<false,false,true,false,false><<<g,128,0,stream>>>(
                XB, wql, nullptr, QB, BTv, Cv, Cv, Cv, Cv, Cv,
                1, 0,0, 0,0, 0,0, 1.f);
            k_gemm<false,false,true,false,false><<<g,128,0,stream>>>(
                XB, wkl, nullptr, KB, BTv, Cv, Cv, Cv, Cv, Cv,
                1, 0,0, 0,0, 0,0, 1.f);
            k_gemm<false,false,true,false,true><<<g,128,0,stream>>>(
                XB, wvl, nullptr, VT, BTv, Cv, Cv, Cv, Cv, Cv,
                1, 0,0, 0,0, 0,0, 1.f);
        }
        // scores = (Qh x Kh^T) / 8, causal groups only, bf16 [B,H,T,T]
        {
            dim3 g(blocks_for(Tv, Tv), 1, Bv * Hv);
            k_gemm<false,false,true,true,false><<<g,128,0,stream>>>(
                QB, KB, nullptr, SC, Tv, Tv, HSv, Cv, Cv, Tv,
                Hv,
                (long long)Tv * Cv, HSv,
                (long long)Tv * Cv, HSv,
                (long long)Hv * Tv * Tv, (long long)Tv * Tv,
                0.125f);
        }
        // causal softmax in place
        k_softmax<<<Bv * Hv * Tv, 256, 0, stream>>>(SC);
        // attn = P x Vh -> AT [BT, C] (head-interleaved columns); B = VT is K-contiguous
        {
            dim3 g(blocks_for(Tv, HSv), 1, Bv * Hv);
            k_gemm<false,false,true,false,false><<<g,128,0,stream>>>(
                SC, VT, nullptr, AT, Tv, HSv, Tv, Tv, Tv, Cv,
                Hv,
                (long long)Hv * Tv * Tv, (long long)Tv * Tv,
                (long long)Cv * Tv, (long long)HSv * Tv,
                (long long)Tv * Cv, HSv,
                1.f);
        }
        // sa = AT x Wproj + bproj -> f32 SA
        {
            dim3 g(blocks_for(BTv, Cv), 1, 1);
            k_gemm<true,false,false,false,false><<<g,128,0,stream>>>(
                AT, wpl, bproj + (size_t)l * Cv, SA, BTv, Cv, Cv, Cv, Cv, Cv,
                1, 0,0, 0,0, 0,0, 1.f);
        }
        // x += LN(sa) ; XB = bf16(x)
        k_ln<true><<<BTv, 256, 0, stream>>>(SA, ln1g + (size_t)l * Cv, ln1b + (size_t)l * Cv, XF, XB);
        // h = relu(XB x W1 + b1) -> bf16 SC [BT, DFF]
        {
            dim3 g(blocks_for(BTv, DFFv), 1, 1);
            k_gemm<true,true,true,false,false><<<g,128,0,stream>>>(
                XB, w1l, b1 + (size_t)l * DFFv, SC, BTv, DFFv, Cv, Cv, Cv, DFFv,
                1, 0,0, 0,0, 0,0, 1.f);
        }
        // ffn = h x W2 + b2 -> f32 SA
        {
            dim3 g(blocks_for(BTv, Cv), 1, 1);
            k_gemm<true,false,false,false,false><<<g,128,0,stream>>>(
                SC, w2l, b2 + (size_t)l * Cv, SA, BTv, Cv, DFFv, DFFv, DFFv, Cv,
                1, 0,0, 0,0, 0,0, 1.f);
        }
        // x += LN(ffn) ; XB = bf16(x)
        k_ln<true><<<BTv, 256, 0, stream>>>(SA, ln2g + (size_t)l * Cv, ln2b + (size_t)l * Cv, XF, XB);
    }

    // final LN -> bf16 in AT
    k_ln<false><<<BTv, 256, 0, stream>>>(XF, lnfg, lnfb, nullptr, AT);

    // logits = AT x Wlm + blm -> f32 d_out [BT, V]
    {
        dim3 g(blocks_for(BTv, Vv), 1, 1);
        k_gemm<true,false,false,false,false><<<g,128,0,stream>>>(
            AT, WLt, blm, out, BTv, Vv, Cv, Cv, Cv, Vv,
            1, 0,0, 0,0, 0,0, 1.f);
    }

    // cross-entropy loss -> d_out[BT*V]
    k_rowloss<<<BTv, 128, 0, stream>>>(out, target, RL);
    k_loss_reduce<<<1, 256, 0, stream>>>(RL, out + (size_t)BTv * Vv);
}
